// SelfAttentionLayerSparse_13606456394290
// MI455X (gfx1250) — matrix operand order repaired and checked by reading, not compiled, not run on hardware
//
#include <hip/hip_runtime.h>
#include <stdint.h>

// ---- problem constants (match reference) ----
#define NNODES 20000
#define NEDGES 250000
#define FIN    512
#define FQK    512
#define FV     512
#define NH     8
#define FTOT   (2*FQK + FV)   // 1536
#define FH     (FQK / NH)     // 64
#define SCALE  0.125f         // FH^-0.5

// ---- CDNA5 WMMA vector types ----
typedef __attribute__((ext_vector_type(16))) __bf16 v16bf;
typedef __attribute__((ext_vector_type(8)))  __bf16 v8bf;
typedef __attribute__((ext_vector_type(8)))  float  v8f;

// f32 -> bf16, round-to-nearest-even (manual, header-independent)
__device__ __forceinline__ uint16_t f2bf(float f) {
    unsigned u = __float_as_uint(f);
    unsigned r = 0x7FFFu + ((u >> 16) & 1u);
    return (uint16_t)((u + r) >> 16);
}

// float atomic max via signed/unsigned ordering trick (init must be -inf)
__device__ __forceinline__ void atomicMaxF32(float* addr, float val) {
    if (val >= 0.0f) atomicMax((int*)addr, __float_as_int(val));
    else             atomicMin((unsigned*)addr, __float_as_uint(val));
}

// ======================= conversion kernels =======================
__global__ void cvt_x_kernel(const float* __restrict__ x, uint16_t* __restrict__ xb) {
    int idx = blockIdx.x * blockDim.x + threadIdx.x;
    if (idx < NNODES * FIN) xb[idx] = f2bf(x[idx]);
}

// W is [FIN][FTOT]; store transposed bf16 Wt[FTOT][FIN] so B fragments are contiguous
__global__ void cvt_w_kernel(const float* __restrict__ W, uint16_t* __restrict__ wt) {
    int idx = blockIdx.x * blockDim.x + threadIdx.x;
    if (idx >= FIN * FTOT) return;
    int k = idx / FTOT;
    int n = idx % FTOT;
    wt[(size_t)n * FIN + k] = f2bf(W[idx]);
}

__global__ void init_out_kernel(float* __restrict__ out) {
    int idx = blockIdx.x * blockDim.x + threadIdx.x;
    if (idx < NNODES * FV) out[idx] = 0.0f;
}

__global__ void init_nh_kernel(float* __restrict__ mmax, float* __restrict__ denom) {
    int idx = blockIdx.x * blockDim.x + threadIdx.x;
    if (idx < NNODES * NH) { mmax[idx] = -__builtin_inff(); denom[idx] = 0.0f; }
}

// ======================= WMMA GEMM: qkv = x @ W =======================
#define BM 128
#define BN 128
#define BK 32

__global__ __launch_bounds__(256) void gemm_qkv_kernel(
    const uint16_t* __restrict__ xb,   // [NNODES][FIN] bf16
    const uint16_t* __restrict__ wt,   // [FTOT][FIN]   bf16 (W transposed)
    float* __restrict__ qkv)           // [NNODES][FTOT] f32
{
    __shared__ uint16_t As[BM][BK];    // A tile: rows of x
    __shared__ uint16_t Bs[BN][BK];    // B tile: columns of W (rows of Wt)

    const int t    = threadIdx.x;
    const int wid  = t >> 5;
    const int lane = t & 31;
    const int lrow = lane & 15;   // M index (A/C) or N index (B/C)
    const int lhi  = lane >> 4;   // K-half selector (A/B), M+8 selector (C)
    const int wm   = wid & 3;     // wave row   (4 waves tall)
    const int wn   = wid >> 2;    // wave col   (2 waves wide)

    const int m0 = blockIdx.x * BM;
    const int n0 = blockIdx.y * BN;

    // staging: each thread copies 16 bf16 (32B) of one LDS row
    const int srow  = t >> 1;     // 0..127
    const int shalf = t & 1;      // which 16-element half of the 32-K row

    v8f vzero = {};
    v8f acc[2][4];
    #pragma unroll
    for (int i = 0; i < 2; ++i)
        #pragma unroll
        for (int j = 0; j < 4; ++j) acc[i][j] = vzero;

    for (int k0 = 0; k0 < FIN; k0 += BK) {
        // ---- stage A tile (clamp rows past NNODES) ----
        int gm = m0 + srow; if (gm >= NNODES) gm = NNODES - 1;
        const uint4* ga = reinterpret_cast<const uint4*>(xb + (size_t)gm * FIN + k0 + shalf * 16);
        uint4* la = reinterpret_cast<uint4*>(&As[srow][shalf * 16]);
        la[0] = ga[0]; la[1] = ga[1];
        // ---- stage B tile (FTOT divisible by BN, no clamp) ----
        const uint4* gb = reinterpret_cast<const uint4*>(wt + (size_t)(n0 + srow) * FIN + k0 + shalf * 16);
        uint4* lb = reinterpret_cast<uint4*>(&Bs[srow][shalf * 16]);
        lb[0] = gb[0]; lb[1] = gb[1];
        __syncthreads();

        // ---- assemble fragments per ISA 16-bit A/B layouts ----
        union Frag { v16bf v; v8bf h[2]; };
        Frag fa[2], fb[4];
        #pragma unroll
        for (int i = 0; i < 2; ++i) {
            // A 16x32: lane m=lrow; VGPR0-3: K=(lhi*8)..+7, VGPR4-7: K=+16..+23
            const uint16_t* pa = &As[wm * 32 + i * 16 + lrow][lhi * 8];
            fa[i].h[0] = *reinterpret_cast<const v8bf*>(pa);
            fa[i].h[1] = *reinterpret_cast<const v8bf*>(pa + 16);
        }
        #pragma unroll
        for (int j = 0; j < 4; ++j) {
            // B 32x16: lane n=lrow; VGPR0-3: K=(lhi*8)..+7, VGPR4-7: K=+16..+23
            const uint16_t* pb = &Bs[wn * 64 + j * 16 + lrow][lhi * 8];
            fb[j].h[0] = *reinterpret_cast<const v8bf*>(pb);
            fb[j].h[1] = *reinterpret_cast<const v8bf*>(pb + 16);
        }

        // ---- 8 WMMAs per wave per K-step ----
        #pragma unroll
        for (int i = 0; i < 2; ++i)
            #pragma unroll
            for (int j = 0; j < 4; ++j)
                acc[i][j] = __builtin_amdgcn_wmma_f32_16x16x32_bf16(
                    false, fa[i].v, false, fb[j].v,
                    (short)0, acc[i][j], false, false);
        __syncthreads();
    }

    // ---- store C per ISA layout: VGPR r -> M = r + lhi*8, N = lrow ----
    #pragma unroll
    for (int i = 0; i < 2; ++i) {
        #pragma unroll
        for (int j = 0; j < 4; ++j) {
            const int col = n0 + wn * 64 + j * 16 + lrow;
            #pragma unroll
            for (int r = 0; r < 8; ++r) {
                const int m = m0 + wm * 32 + i * 16 + lhi * 8 + r;
                if (m < NNODES) qkv[(size_t)m * FTOT + col] = acc[i][j][r];
            }
        }
    }
}

// ======================= edge phase =======================
// pass 1: per-(edge,head) logits + segment max
__global__ void edge_scores_kernel(const float* __restrict__ qkv,
                                   const int* __restrict__ ei,
                                   float* __restrict__ aw,
                                   float* __restrict__ mmax)
{
    int idx = blockIdx.x * blockDim.x + threadIdx.x;
    if (idx >= NEDGES * NH) return;
    int e = idx >> 3, h = idx & 7;
    int s = ei[e], d = ei[NEDGES + e];
    const float4* q = reinterpret_cast<const float4*>(qkv + (size_t)s * FTOT + h * FH);
    const float4* k = reinterpret_cast<const float4*>(qkv + (size_t)d * FTOT + FQK + h * FH);
    float acc = 0.0f;
    #pragma unroll
    for (int i = 0; i < FH / 4; ++i) {
        float4 a = q[i], b = k[i];
        acc += a.x * b.x + a.y * b.y + a.z * b.z + a.w * b.w;
    }
    acc *= SCALE;
    aw[idx] = acc;
    atomicMaxF32(&mmax[s * NH + h], acc);
}

// pass 2: exp(s - m) + segment sum
__global__ void edge_exp_kernel(const int* __restrict__ ei,
                                float* __restrict__ aw,
                                const float* __restrict__ mmax,
                                float* __restrict__ denom)
{
    int idx = blockIdx.x * blockDim.x + threadIdx.x;
    if (idx >= NEDGES * NH) return;
    int e = idx >> 3, h = idx & 7;
    int s = ei[e];
    float v = __expf(aw[idx] - mmax[s * NH + h]);
    aw[idx] = v;
    atomicAdd(&denom[s * NH + h], v);
}

// pass 3: alpha = e/denom; out[src] += alpha * v[dest]
__global__ void edge_scatter_kernel(const float* __restrict__ qkv,
                                    const int* __restrict__ ei,
                                    const float* __restrict__ aw,
                                    const float* __restrict__ denom,
                                    float* __restrict__ out)
{
    int idx = blockIdx.x * blockDim.x + threadIdx.x;
    if (idx >= NEDGES * NH) return;
    int e = idx >> 3, h = idx & 7;
    int s = ei[e], d = ei[NEDGES + e];
    float alpha = aw[idx] / denom[s * NH + h];
    const float4* v = reinterpret_cast<const float4*>(qkv + (size_t)d * FTOT + 2 * FQK + h * FH);
    float* o = out + (size_t)s * FV + h * FH;
    #pragma unroll
    for (int i = 0; i < FH / 4; ++i) {
        float4 vv = v[i];
        atomicAdd(&o[4 * i + 0], alpha * vv.x);
        atomicAdd(&o[4 * i + 1], alpha * vv.y);
        atomicAdd(&o[4 * i + 2], alpha * vv.z);
        atomicAdd(&o[4 * i + 3], alpha * vv.w);
    }
}

// ======================= host launcher =======================
extern "C" void kernel_launch(void* const* d_in, const int* in_sizes, int n_in,
                              void* d_out, int out_size, void* d_ws, size_t ws_size,
                              hipStream_t stream) {
    (void)in_sizes; (void)n_in; (void)out_size; (void)ws_size;
    const float* x  = (const float*)d_in[0];
    // d_in[1] = batch (unused by the reference computation)
    const int*   ei = (const int*)d_in[2];
    const float* W  = (const float*)d_in[3];
    float* out = (float*)d_out;

    // workspace layout (all segments 256B-aligned by construction)
    char* p = (char*)d_ws;
    float*    qkv   = (float*)p;    p += (size_t)NNODES * FTOT * sizeof(float);   // 122.88 MB
    float*    aw    = (float*)p;    p += (size_t)NEDGES * NH   * sizeof(float);   //   8.00 MB
    float*    mmax  = (float*)p;    p += (size_t)NNODES * NH   * sizeof(float);   //   0.64 MB
    float*    denom = (float*)p;    p += (size_t)NNODES * NH   * sizeof(float);   //   0.64 MB
    uint16_t* xb    = (uint16_t*)p; p += (size_t)NNODES * FIN  * sizeof(uint16_t);// 20.48 MB
    uint16_t* wt    = (uint16_t*)p;                                               //  1.57 MB

    const int TB = 256;
    cvt_x_kernel   <<<(NNODES * FIN + TB - 1) / TB, TB, 0, stream>>>(x, xb);
    cvt_w_kernel   <<<(FIN * FTOT + TB - 1) / TB, TB, 0, stream>>>(W, wt);
    init_out_kernel<<<(NNODES * FV + TB - 1) / TB, TB, 0, stream>>>(out);
    init_nh_kernel <<<(NNODES * NH + TB - 1) / TB, TB, 0, stream>>>(mmax, denom);

    dim3 gg((NNODES + BM - 1) / BM, FTOT / BN);   // (157, 12)
    gemm_qkv_kernel<<<gg, TB, 0, stream>>>(xb, wt, qkv);

    const int eg = (NEDGES * NH + TB - 1) / TB;
    edge_scores_kernel <<<eg, TB, 0, stream>>>(qkv, ei, aw, mmax);
    edge_exp_kernel    <<<eg, TB, 0, stream>>>(ei, aw, mmax, denom);
    edge_scatter_kernel<<<eg, TB, 0, stream>>>(qkv, ei, aw, denom, out);
}
